// GruAutoEncoder_19980187861881
// MI455X (gfx1250) — matrix operand order repaired and checked by reading, not compiled, not run on hardware
//
#include <hip/hip_runtime.h>
#include <hip/hip_bf16.h>
#include <math.h>

typedef __attribute__((ext_vector_type(16))) _Float16 v16h;
typedef __attribute__((ext_vector_type(8)))  float    v8f;
typedef __attribute__((ext_vector_type(4)))  unsigned int u32x4;
typedef __attribute__((ext_vector_type(4)))  int      i32x4;
typedef __attribute__((ext_vector_type(8)))  int      i32x8;

#define B_  128
#define T_  512
#define E_  256
#define H_  256
#define G3  768                       // 3*H
#define NT_ 48                        // G3/16 column tiles
#define KT_ 8                         // E/32 k tiles
#define FRAG_HALVES (NT_*KT_*32*16)   // 196608 halves per weight matrix

union AFrag { v16h v; unsigned int u[8]; };

__device__ __forceinline__ v8f wmma16(v16h a, v16h b, v8f c) {
  // emits v_wmma_f32_16x16x32_f16
  return __builtin_amdgcn_wmma_f32_16x16x32_f16(false, a, false, b, (short)0, c, false, false);
}

// ---------------------------------------------------------------------------
// K0: shuffle 4 weight matrices [768,256] f32 -> f16 WMMA B-fragment layout.
// ---------------------------------------------------------------------------
__global__ void __launch_bounds__(256) shuffle_weights(
    const float* __restrict__ w0, const float* __restrict__ w1,
    const float* __restrict__ w2, const float* __restrict__ w3,
    _Float16* __restrict__ dst) {
  int idx = blockIdx.x * 256 + threadIdx.x;
  if (idx >= 4 * FRAG_HALVES) return;
  int w   = idx / FRAG_HALVES;
  int rem = idx - w * FRAG_HALVES;
  int nt = rem >> 12;
  int r2 = rem & 4095;
  int kt = r2 >> 9;
  int r3 = r2 & 511;
  int ln = r3 >> 4;
  int hh = r3 & 15;
  int j = nt * 16 + (ln & 15);
  int k = kt * 32 + (hh >> 3) * 16 + (ln >> 4) * 8 + (hh & 7);
  const float* W = (w == 0) ? w0 : (w == 1) ? w1 : (w == 2) ? w2 : w3;
  dst[idx] = (_Float16)W[j * E_ + k];
}

// ---------------------------------------------------------------------------
// K0b: convert x [B*T,E] f32 -> f16 (row-major, contiguous) so the TDM can
// stage raw f16 tiles. 4 elements / thread.
// ---------------------------------------------------------------------------
__global__ void __launch_bounds__(256) convert_x(const float* __restrict__ x,
                                                 _Float16* __restrict__ xh) {
  size_t i = ((size_t)blockIdx.x * 256 + threadIdx.x) * 4;
  const float4* xv = (const float4*)(x + i);
  float4 v = *xv;
  union { _Float16 h[4]; unsigned long long q; } o;
  o.h[0] = (_Float16)v.x; o.h[1] = (_Float16)v.y;
  o.h[2] = (_Float16)v.z; o.h[3] = (_Float16)v.w;
  *(unsigned long long*)(xh + i) = o.q;
}

// ---------------------------------------------------------------------------
// K1: Xg[65536,768] = x[65536,256] @ Wih_e^T   (raw, biases added later)
// A-tile (64x256 f16, padded to 264 halves/row) staged by the Tensor Data
// Mover: D# pad feature inserts 4 dwords after every 128 dwords (one row).
// ---------------------------------------------------------------------------
__global__ void __launch_bounds__(256) xin_gemm(
    const _Float16* __restrict__ xh, const _Float16* __restrict__ wf,
    float* __restrict__ xg) {
  __shared__ _Float16 xs[64][264];   // 33 KB, 132 dwords/row
  int tid = threadIdx.x;
  int wgm = blockIdx.x >> 3;
  int wgn = blockIdx.x & 7;

  if (tid < 32) {   // one wave issues the TDM DMA (EXEC ignored, per-wave op)
    unsigned long long ga =
        (unsigned long long)(uintptr_t)(xh + (size_t)wgm * 64 * E_);
    unsigned lds_base = (unsigned)(uintptr_t)&xs[0][0];
    u32x4 g0;
    g0.x = 1u;                                        // count=1 (valid)
    g0.y = lds_base;                                  // lds_addr
    g0.z = (unsigned)(ga & 0xFFFFFFFFull);            // global_addr[31:0]
    g0.w = (unsigned)((ga >> 32) & 0x01FFFFFFull) | (2u << 30); // type=2
    i32x8 g1;
    g1[0] = (1 << 16)            // data_size = 2 bytes
          | (1 << 20)            // pad_enable
          | (6 << 22)            // pad_interval: 128 dwords (one 512B row)
          | (3 << 25);           // pad_amount: 4 dwords (16B) -> 264-half rows
    g1[1] = (int)(256u << 16);   // tensor_dim0 = 256 (low16 in bits 63:48)
    g1[2] = 0;                   // tensor_dim0 hi16=0, tensor_dim1 low16=0
    g1[3] = 1 | (256 << 16);     // tensor_dim1 = 65536, tile_dim0 = 256
    g1[4] = 64;                  // tile_dim1 = 64 rows, tile_dim2 = 0
    g1[5] = 256;                 // tensor_dim0_stride = 256 elements
    g1[6] = 0;
    g1[7] = 0;
    i32x4 z4 = {0, 0, 0, 0};
#if defined(__clang_major__) && (__clang_major__ >= 23)
    i32x8 z8 = {0, 0, 0, 0, 0, 0, 0, 0};
    __builtin_amdgcn_tensor_load_to_lds(g0, g1, z4, z4, z8, 0);
#else
    __builtin_amdgcn_tensor_load_to_lds(g0, g1, z4, z4, 0);
#endif
    __builtin_amdgcn_s_wait_tensorcnt(0);
  }
  __syncthreads();

  int lane = tid & 31, wv = tid >> 5;
  int lo = lane & 15, grp = lane >> 4;
  int mt = wv & 3, nset = wv >> 2;

  const unsigned int* xrow = (const unsigned int*)&xs[mt * 16 + lo][0];
  AFrag a[KT_];
#pragma unroll
  for (int kt = 0; kt < KT_; ++kt)
#pragma unroll
    for (int v = 0; v < 4; ++v) {
      a[kt].u[v]     = xrow[kt * 16 + grp * 4 + v];
      a[kt].u[4 + v] = xrow[kt * 16 + 8 + grp * 4 + v];
    }

  const v16h* bf = (const v16h*)wf;
#pragma unroll
  for (int i = 0; i < 3; ++i) {
    int ntg = wgn * 6 + nset * 3 + i;
    v8f acc = {};
#pragma unroll
    for (int kt = 0; kt < KT_; ++kt)
      acc = wmma16(a[kt].v, bf[(ntg * KT_ + kt) * 32 + lane], acc);
    float* orow = xg + (size_t)(wgm * 64 + mt * 16 + grp * 8) * G3 + ntg * 16 + lo;
#pragma unroll
    for (int r = 0; r < 8; ++r)
      orow[(size_t)r * G3] = acc[r];
  }
}

// ---------------------------------------------------------------------------
// K2: encoder recurrence. 8 WGs x 16 batch rows, 512 steps in-kernel.
// ---------------------------------------------------------------------------
__global__ void __launch_bounds__(512) gru_encode(
    const float* __restrict__ xg, const _Float16* __restrict__ whhf,
    const float* __restrict__ bih, const float* __restrict__ bhh,
    float* __restrict__ lat_out, _Float16* __restrict__ lat_h) {
  __shared__ _Float16 hbuf[16][264];
  __shared__ float    gg[16][776];
  int tid = threadIdx.x;
  int b0 = blockIdx.x * 16;

  for (int i = tid; i < 16 * 264; i += 512) ((_Float16*)hbuf)[i] = (_Float16)0.f;
  __syncthreads();

  int lane = tid & 31, wv = tid >> 5;          // 16 waves, 3 n-tiles each
  int lo = lane & 15, grp = lane >> 4;
  int j = tid & 255, mb = (tid >> 8) * 8;
  const v16h* bf = (const v16h*)whhf;
  float bihr = bih[j], bihz = bih[j + 256], bihn = bih[j + 512];
  float bhr = bhh[j], bhz = bhh[j + 256], bhn = bhh[j + 512];

  for (int t = 0; t < T_; ++t) {
    const unsigned int* hrow = (const unsigned int*)&hbuf[lo][0];
    AFrag a[KT_];
#pragma unroll
    for (int kt = 0; kt < KT_; ++kt)
#pragma unroll
      for (int v = 0; v < 4; ++v) {
        a[kt].u[v]     = hrow[kt * 16 + grp * 4 + v];
        a[kt].u[4 + v] = hrow[kt * 16 + 8 + grp * 4 + v];
      }
#pragma unroll
    for (int i = 0; i < 3; ++i) {
      int nt = wv * 3 + i;
      v8f acc = {};
#pragma unroll
      for (int kt = 0; kt < KT_; ++kt)
        acc = wmma16(a[kt].v, bf[(nt * KT_ + kt) * 32 + lane], acc);
#pragma unroll
      for (int r = 0; r < 8; ++r) gg[grp * 8 + r][nt * 16 + lo] = acc[r];
    }
    __syncthreads();

#pragma unroll
    for (int mm = 0; mm < 8; ++mm) {
      int m = mb + mm;
      const float* xr = xg + ((size_t)(b0 + m) * T_ + t) * G3;
      if (t + 1 < T_) __builtin_prefetch(xr + G3 + j, 0, 1); // next-step Xg row
      float gir = xr[j] + bihr, giz = xr[j + 256] + bihz, gin = xr[j + 512] + bihn;
      float ghr = gg[m][j] + bhr, ghz = gg[m][j + 256] + bhz, ghn = gg[m][j + 512] + bhn;
      float rg = 1.f / (1.f + expf(-(gir + ghr)));
      float zg = 1.f / (1.f + expf(-(giz + ghz)));
      float ng = tanhf(gin + rg * ghn);
      float hp = (float)hbuf[m][j];
      float hn = (1.f - zg) * ng + zg * hp;
      hbuf[m][j] = (_Float16)hn;
      if (t == T_ - 1) {
        lat_out[(size_t)(b0 + m) * H_ + j] = hn;
        lat_h[(b0 + m) * H_ + j] = (_Float16)hn;
      }
    }
    __syncthreads();
  }
}

// ---------------------------------------------------------------------------
// K3: decoder recurrence. input==state==y_prev (step 0: input=latents,
// state=0 handled via scale s). Two GEMM phases share one gate buffer.
// ---------------------------------------------------------------------------
__global__ void __launch_bounds__(512) gru_decode(
    const _Float16* __restrict__ lat_h,
    const _Float16* __restrict__ wihf, const _Float16* __restrict__ whhf,
    const float* __restrict__ bih, const float* __restrict__ bhh,
    float* __restrict__ out) {
  __shared__ _Float16 ybuf[16][264];
  __shared__ float    gg[16][776];
  int tid = threadIdx.x;
  int b0 = blockIdx.x * 16;

  for (int i = tid; i < 16 * 256; i += 512) {
    int m = i >> 8, c = i & 255;
    ybuf[m][c] = lat_h[(b0 + m) * H_ + c];
  }
  __syncthreads();

  int lane = tid & 31, wv = tid >> 5;
  int lo = lane & 15, grp = lane >> 4;
  int j = tid & 255, mb = (tid >> 8) * 8;
  const v16h* bfi = (const v16h*)wihf;
  const v16h* bfh = (const v16h*)whhf;
  float bir = bih[j], biz = bih[j + 256], binn = bih[j + 512];
  float bhr = bhh[j], bhz = bhh[j + 256], bhn = bhh[j + 512];

  for (int t = 0; t < T_; ++t) {
    float s = (t == 0) ? 0.f : 1.f;   // state is zero on first step
    const unsigned int* yrow = (const unsigned int*)&ybuf[lo][0];
    AFrag a[KT_];
#pragma unroll
    for (int kt = 0; kt < KT_; ++kt)
#pragma unroll
      for (int v = 0; v < 4; ++v) {
        a[kt].u[v]     = yrow[kt * 16 + grp * 4 + v];
        a[kt].u[4 + v] = yrow[kt * 16 + 8 + grp * 4 + v];
      }

    // phase A: gi = y @ Wih_d^T
#pragma unroll
    for (int i = 0; i < 3; ++i) {
      int nt = wv * 3 + i;
      v8f acc = {};
#pragma unroll
      for (int kt = 0; kt < KT_; ++kt)
        acc = wmma16(a[kt].v, bfi[(nt * KT_ + kt) * 32 + lane], acc);
#pragma unroll
      for (int r = 0; r < 8; ++r) gg[grp * 8 + r][nt * 16 + lo] = acc[r];
    }
    __syncthreads();

    float gir[8], giz[8], gin[8];
#pragma unroll
    for (int mm = 0; mm < 8; ++mm) {
      int m = mb + mm;
      gir[mm] = gg[m][j] + bir;
      giz[mm] = gg[m][j + 256] + biz;
      gin[mm] = gg[m][j + 512] + binn;
    }
    __syncthreads();

    // phase B: gh = state @ Whh_d^T (same A fragments)
#pragma unroll
    for (int i = 0; i < 3; ++i) {
      int nt = wv * 3 + i;
      v8f acc = {};
#pragma unroll
      for (int kt = 0; kt < KT_; ++kt)
        acc = wmma16(a[kt].v, bfh[(nt * KT_ + kt) * 32 + lane], acc);
#pragma unroll
      for (int r = 0; r < 8; ++r) gg[grp * 8 + r][nt * 16 + lo] = acc[r];
    }
    __syncthreads();

#pragma unroll
    for (int mm = 0; mm < 8; ++mm) {
      int m = mb + mm;
      float ghr = s * gg[m][j] + bhr;
      float ghz = s * gg[m][j + 256] + bhz;
      float ghn = s * gg[m][j + 512] + bhn;
      float rg = 1.f / (1.f + expf(-(gir[mm] + ghr)));
      float zg = 1.f / (1.f + expf(-(giz[mm] + ghz)));
      float ng = tanhf(gin[mm] + rg * ghn);
      float yp = s * (float)ybuf[m][j];
      float yn = (1.f - zg) * ng + zg * yp;
      out[((size_t)(b0 + m) * T_ + (T_ - 1 - t)) * H_ + j] = yn;   // flipped time
      ybuf[m][j] = (_Float16)yn;
    }
    __syncthreads();
  }
}

// ---------------------------------------------------------------------------
extern "C" void kernel_launch(void* const* d_in, const int* in_sizes, int n_in,
                              void* d_out, int out_size, void* d_ws, size_t ws_size,
                              hipStream_t stream) {
  (void)in_sizes; (void)n_in; (void)out_size; (void)ws_size;
  const float* x     = (const float*)d_in[0];
  const float* Wih_e = (const float*)d_in[1];
  const float* Whh_e = (const float*)d_in[2];
  const float* bih_e = (const float*)d_in[3];
  const float* bhh_e = (const float*)d_in[4];
  const float* Wih_d = (const float*)d_in[5];
  const float* Whh_d = (const float*)d_in[6];
  const float* bih_d = (const float*)d_in[7];
  const float* bhh_d = (const float*)d_in[8];
  float* out = (float*)d_out;

  char* ws = (char*)d_ws;
  _Float16* fW     = (_Float16*)ws;                 // 4 weight frag arrays
  _Float16* fWih_e = fW;
  _Float16* fWhh_e = fW + FRAG_HALVES;
  _Float16* fWih_d = fW + 2 * FRAG_HALVES;
  _Float16* fWhh_d = fW + 3 * FRAG_HALVES;
  size_t off = (size_t)4 * FRAG_HALVES * 2;
  _Float16* xh = (_Float16*)(ws + off);             // x in f16, 33.5 MB
  off += (size_t)B_ * T_ * E_ * 2;
  float* xg = (float*)(ws + off);                   // Xg raw, 201 MB
  off += (size_t)B_ * T_ * G3 * 4;
  _Float16* lat_h = (_Float16*)(ws + off);
  float* lat_out = out + (size_t)B_ * T_ * H_;

  shuffle_weights<<<(4 * FRAG_HALVES + 255) / 256, 256, 0, stream>>>(
      Wih_e, Whh_e, Wih_d, Whh_d, fW);
  convert_x<<<(B_ * T_ * E_ / 4) / 256, 256, 0, stream>>>(x, xh);
  xin_gemm<<<(B_ * T_ / 64) * 8, 256, 0, stream>>>(xh, fWih_e, xg);
  gru_encode<<<B_ / 16, 512, 0, stream>>>(xg, fWhh_e, bih_e, bhh_e, lat_out, lat_h);
  gru_decode<<<B_ / 16, 512, 0, stream>>>(lat_h, fWih_d, fWhh_d, bih_d, bhh_d, out);
}
